// BiModalAttention_6270652252304
// MI455X (gfx1250) — compile-verified, matrix operand order checked by reading, not run on hardware
//
#include <hip/hip_runtime.h>

#define B_DIM 8
#define S_DIM 4096
#define D_DIM 64
#define BLOCK_N 64
#define WAVES 8
#define KV_STRIDE 72   // halves: 64 + 8 pad
#define VT_STRIDE 72
#define P_STRIDE  72
#define LOG2E 1.44269504088896340736f

typedef __attribute__((ext_vector_type(16))) _Float16 v16h;
typedef __attribute__((ext_vector_type(8)))  _Float16 v8h;
typedef __attribute__((ext_vector_type(4)))  _Float16 v4h;
typedef __attribute__((ext_vector_type(8)))  float    v8f;

// 16-bit matrix fragment (16x32), chunk swizzle per ISA A-table (B assumed
// symmetric): halves 0..7 -> K = 8*hi + h ; halves 8..15 -> K = 16 + 8*hi + h-8
__device__ __forceinline__ v16h ld_frag(const _Float16* row, int hi) {
    v8h lo = *(const v8h*)(row + 8 * hi);
    v8h hh = *(const v8h*)(row + 16 + 8 * hi);
    return __builtin_shufflevector(lo, hh, 0,1,2,3,4,5,6,7,8,9,10,11,12,13,14,15);
}

// Q^T B-fragment straight from global f32, pre-scaled by log2(e)
__device__ __forceinline__ v16h ld_qfrag_global(const float* row, int hi) {
    v16h a;
#pragma unroll
    for (int i = 0; i < 8; ++i) a[i] = (_Float16)(row[8 * hi + i] * LOG2E);
#pragma unroll
    for (int i = 0; i < 8; ++i) a[8 + i] = (_Float16)(row[16 + 8 * hi + i] * LOG2E);
    return a;
}

__device__ __forceinline__ v8f wmma16(v16h a, v16h b, v8f c) {
    return __builtin_amdgcn_wmma_f32_16x16x32_f16(false, a, false, b, (short)0, c,
                                                  false, false);
}

// ---------------------------------------------------------------------------
// One block = (dir, batch, 128 query rows); each wave owns 16 queries.
// Transposed flash attention: S^T = K.Q^T, o^T = V^T.P^T. Per-query softmax
// stats live per-lane (single shfl_xor(16) for the max); the running sum is
// computed on the matrix pipe as ones^T . P^T; K/V tile is double-buffered.
__global__ __launch_bounds__(WAVES * 32)
void bimodal_attn_kernel(const float* __restrict__ x,
                         const float* __restrict__ y,
                         float* __restrict__ out) {
    __shared__ __align__(16) _Float16 s_kv[2][BLOCK_N * KV_STRIDE]; // K rows (k-major)
    __shared__ __align__(16) _Float16 s_vt[2][D_DIM * VT_STRIDE];   // V^T rows (d-major)
    __shared__ __align__(16) _Float16 s_p[WAVES][16 * P_STRIDE];    // P^T per wave [m][kappa]

    const int dir   = blockIdx.z;
    const int b     = blockIdx.y;
    const int qtile = blockIdx.x;
    const int tid   = threadIdx.x;
    const int wv    = tid >> 5;
    const int lane  = tid & 31;
    const int hi    = lane >> 4;
    const int lr    = lane & 15;

    const float* Q  = dir ? y : x;
    const float* KV = dir ? x : y;

    // ---- Q^T B-fragments, once (this lane's query row, scaled by log2e) ----
    const size_t qrow_g = (size_t)b * S_DIM + qtile * 128 + wv * 16 + lr;
    const float* qptr = Q + qrow_g * D_DIM;
    const v16h qb0 = ld_qfrag_global(qptr, hi);
    const v16h qb1 = ld_qfrag_global(qptr + 32, hi);

    // ---- all-ones A fragment: ones^T . P^T == per-query probability sum ----
    v16h ones_a;
#pragma unroll
    for (int i = 0; i < 16; ++i) ones_a[i] = (_Float16)1.0f;

    // ---- per-lane flash state (query m = lr, mirrored in both hi halves) ----
    float Mrun = -1e30f, Lrun = 0.0f;
    v8f o[4];
#pragma unroll
    for (int j = 0; j < 4; ++j) o[j] = (v8f)(0.0f);

    // ---- staging: 4x4 micro-transpose per thread, 64x64 f32 tile ----
    const int kr0 = (tid >> 4) * 4;          // kappa group
    const int dc0 = (tid & 15) * 4;          // d group
    const float* kv_base = KV + ((size_t)b * S_DIM + kr0) * D_DIM + dc0;

    float4 f[4];
#pragma unroll
    for (int i = 0; i < 4; ++i) f[i] = *(const float4*)(kv_base + i * D_DIM);

    int p = 0;
    for (int kt = 0; kt < S_DIM; kt += BLOCK_N) {
        // ---- store staged regs to LDS buffer p (row-major + transposed) ----
        _Float16 hv[4][4];
#pragma unroll
        for (int i = 0; i < 4; ++i) {
            hv[i][0] = (_Float16)f[i].x; hv[i][1] = (_Float16)f[i].y;
            hv[i][2] = (_Float16)f[i].z; hv[i][3] = (_Float16)f[i].w;
        }
#pragma unroll
        for (int i = 0; i < 4; ++i) {
            v4h rm = {hv[i][0], hv[i][1], hv[i][2], hv[i][3]};
            *(v4h*)&s_kv[p][(kr0 + i) * KV_STRIDE + dc0] = rm;
        }
#pragma unroll
        for (int j = 0; j < 4; ++j) {
            v4h tp = {hv[0][j], hv[1][j], hv[2][j], hv[3][j]};
            *(v4h*)&s_vt[p][(dc0 + j) * VT_STRIDE + kr0] = tp;
        }
        __syncthreads();

        // ---- issue next tile's global loads (hidden under compute) ----
        if (kt + BLOCK_N < S_DIM) {
            const float* gk = kv_base + (size_t)(kt + BLOCK_N) * D_DIM;
#pragma unroll
            for (int i = 0; i < 4; ++i) f[i] = *(const float4*)(gk + i * D_DIM);
        }
        if (kt + 2 * BLOCK_N < S_DIM)
            __builtin_prefetch(kv_base + (size_t)(kt + 2 * BLOCK_N) * D_DIM, 0, 1);

        // ---- scores: S^T[64k x 16m] = K[64x64] . Q^T, four 16x16 tiles ----
        v8f s[4];
#pragma unroll
        for (int j = 0; j < 4; ++j) {
            const _Float16* krow = &s_kv[p][(16 * j + lr) * KV_STRIDE];
            s[j] = wmma16(ld_frag(krow, hi),      qb0, (v8f)(0.0f));
            s[j] = wmma16(ld_frag(krow + 32, hi), qb1, s[j]);
        }

        // ---- per-lane softmax max: in-register tree + one shfl_xor(16) ----
        float tmax = s[0][0];
#pragma unroll
        for (int j = 0; j < 4; ++j)
#pragma unroll
            for (int r = 0; r < 8; ++r) tmax = fmaxf(tmax, s[j][r]);
        tmax = fmaxf(tmax, __shfl_xor(tmax, 16));
        const float mnew  = fmaxf(Mrun, tmax);
        const float alpha = exp2f(Mrun - mnew);

        // ---- P = exp2(S - m), stored contiguous-kappa per lane ----
        _Float16* prow = &s_p[wv][lr * P_STRIDE + 8 * hi];
#pragma unroll
        for (int j = 0; j < 4; ++j) {
            v8h ph;
#pragma unroll
            for (int r = 0; r < 8; ++r)
                ph[r] = (_Float16)exp2f(s[j][r] - mnew);
            *(v8h*)(prow + 16 * j) = ph;   // kappa = 16j + 8hi + r : contiguous
        }

        // ---- reload P^T as B fragments ----
        const v16h pb0 = ld_frag(&s_p[wv][lr * P_STRIDE], hi);
        const v16h pb1 = ld_frag(&s_p[wv][lr * P_STRIDE + 32], hi);

        // ---- row-sum on the matrix pipe: rs = ones^T . P^T ----
        v8f rs = wmma16(ones_a, pb0, (v8f)(0.0f));
        rs = wmma16(ones_a, pb1, rs);

        // ---- rescale accumulators (independent of rs WMMAs) ----
#pragma unroll
        for (int j = 0; j < 4; ++j) o[j] *= alpha;

        // ---- o^T += V^T[64d x 64k] . P^T[64k x 16m] ----
#pragma unroll
        for (int j2 = 0; j2 < 4; ++j2) {
            const _Float16* vrow = &s_vt[p][(16 * j2 + lr) * VT_STRIDE];
            o[j2] = wmma16(ld_frag(vrow, hi),      pb0, o[j2]);
            o[j2] = wmma16(ld_frag(vrow + 32, hi), pb1, o[j2]);
        }

        Lrun = Lrun * alpha + rs[0];   // every lane holds the full kappa sum
        Mrun = mnew;

        p ^= 1;
    }

    // ---- finalize: a = (o/L) * q, contiguous d per lane -> float4 I/O ----
    const float inv = 1.0f / Lrun;
    float* orow = out + qrow_g * (2 * D_DIM) + dir * D_DIM;
#pragma unroll
    for (int j2 = 0; j2 < 4; ++j2) {
        const int dbase = 16 * j2 + 8 * hi;
        const float4 q0 = *(const float4*)(qptr + dbase);
        const float4 q1 = *(const float4*)(qptr + dbase + 4);
        float4 r0, r1;
        r0.x = o[j2][0] * inv * q0.x;  r0.y = o[j2][1] * inv * q0.y;
        r0.z = o[j2][2] * inv * q0.z;  r0.w = o[j2][3] * inv * q0.w;
        r1.x = o[j2][4] * inv * q1.x;  r1.y = o[j2][5] * inv * q1.y;
        r1.z = o[j2][6] * inv * q1.z;  r1.w = o[j2][7] * inv * q1.w;
        *(float4*)(orow + dbase)     = r0;
        *(float4*)(orow + dbase + 4) = r1;
    }
}

extern "C" void kernel_launch(void* const* d_in, const int* in_sizes, int n_in,
                              void* d_out, int out_size, void* d_ws, size_t ws_size,
                              hipStream_t stream) {
    const float* x = (const float*)d_in[0];
    const float* y = (const float*)d_in[1];
    float* out = (float*)d_out;
    dim3 grid(S_DIM / 128, B_DIM, 2);
    dim3 block(WAVES * 32);
    bimodal_attn_kernel<<<grid, block, 0, stream>>>(x, y, out);
}
